// SimpleCompactnessLoss_45784351375565
// MI455X (gfx1250) — matrix operand order repaired and checked by reading, not compiled, not run on hardware
//
#include <hip/hip_runtime.h>

typedef __attribute__((ext_vector_type(16))) _Float16 v16h;
typedef __attribute__((ext_vector_type(8)))  _Float16 v8h;
typedef __attribute__((ext_vector_type(8)))  float    v8f;
typedef __attribute__((ext_vector_type(4)))  int      v4i;

#define NB 64           // samples per batch staged in LDS
#define DD 256          // embedding dim
#define NCLS 64         // classes
#define F32_STRIDE 260  // padded f32 row stride (floats) -> kills bank conflicts
#define TR_STRIDE 80    // padded f16 transposed row stride (halves), 160B rows (32B-aligned)
#define BLOCK 512       // 16 waves (wave32): wave w owns dim-tile w
#define CEPS 1e-12f

#if defined(__has_builtin)
#if __has_builtin(__builtin_amdgcn_global_load_async_to_lds_b128)
#define HAVE_ASYNC_LDS 1
#endif
#if __has_builtin(__builtin_amdgcn_s_wait_asynccnt)
#define HAVE_WAIT_ASYNC_BUILTIN 1
#endif
#endif

__device__ __forceinline__ void wait_async0() {
#if defined(HAVE_WAIT_ASYNC_BUILTIN)
  __builtin_amdgcn_s_wait_asynccnt(0);
#else
  asm volatile("s_wait_asynccnt 0x0" ::: "memory");
#endif
}

// ---------------- zero workspace ----------------
__global__ void zero_ws_kernel(float* ws, int n) {
  int i = blockIdx.x * blockDim.x + threadIdx.x;
  if (i < n) ws[i] = 0.0f;
}

// ---------------- main streaming + WMMA one-hot segment-sum ----------------
__global__ __launch_bounds__(BLOCK) void compact_main_kernel(
    const float4* __restrict__ emb,     // [N, 256] f32 viewed as float4
    const int*    __restrict__ labels,  // [N] int32
    float* __restrict__ partials,       // [numPartials][64][256]
    float* __restrict__ counts,         // [64]
    int N, int numPartials)
{
  __shared__ float     sF32[NB * F32_STRIDE];        // 66,560 B
  __shared__ _Float16  sT[DD * TR_STRIDE];           // 40,960 B  [dim][sample] f16
  __shared__ _Float16  sA[2 * 4 * 32 * 16];          //  8,192 B  one-hot frags, A-layout
  __shared__ float     sSumSq[NB];
  __shared__ float     sInv[NB];
  __shared__ int       sLab[NB];
  __shared__ float     sCnt[NCLS];

  const int tid  = threadIdx.x;
  const int lane = tid & 31;
  const int w    = tid >> 5;          // wave id == dim tile (0..15)
  const int m    = lane & 15;         // N-col within tile / class-row
  const int hi   = (lane >> 4) & 1;   // lane half

  if (tid < NCLS) sCnt[tid] = 0.0f;

  v8f acc[4];
  #pragma unroll
  for (int cg = 0; cg < 4; ++cg) acc[cg] = (v8f){0,0,0,0,0,0,0,0};

  const int numBatches = N / NB;
  for (int b = blockIdx.x; b < numBatches; b += gridDim.x) {
    __syncthreads();                        // protect LDS reuse (and sCnt init)
    if (tid < NB) sSumSq[tid] = 0.0f;
    __syncthreads();

    const float4* gsrc = emb + (size_t)b * (NB * (DD / 4));

#if defined(HAVE_ASYNC_LDS)
    // --- phase A (async): global -> LDS via async DMA path, no VGPR round-trip ---
    #pragma unroll
    for (int i = 0; i < (NB * DD / 4) / BLOCK; ++i) {   // 8 async b128 per thread
      int idx4 = tid + i * BLOCK;
      int k  = idx4 >> 6;
      int n4 = idx4 & 63;
      __builtin_amdgcn_global_load_async_to_lds_b128(
          (__attribute__((address_space(1))) v4i*)(gsrc + idx4),
          (__attribute__((address_space(3))) v4i*)&sF32[k * F32_STRIDE + n4 * 4],
          0, 0);
    }
    if (tid < NB) {
      int lb = labels[b * NB + tid] & (NCLS - 1);
      sLab[tid] = lb;
      atomicAdd(&sCnt[lb], 1.0f);
    }
    wait_async0();
    __syncthreads();
    {   // sum of squares from LDS: 8 threads per sample, 8 float4 each
      int k = tid & (NB - 1);
      int g = tid >> 6;                 // 0..7
      float ss = 0.0f;
      #pragma unroll
      for (int j = 0; j < 8; ++j) {
        float4 v = *(const float4*)&sF32[k * F32_STRIDE + (g * 8 + j) * 4];
        ss += v.x*v.x + v.y*v.y + v.z*v.z + v.w*v.w;
      }
      atomicAdd(&sSumSq[k], ss);
    }
#else
    // --- phase A (fallback): coalesced sync copy fused with sum-of-squares ---
    #pragma unroll
    for (int i = 0; i < (NB * DD / 4) / BLOCK; ++i) {
      int idx4 = tid + i * BLOCK;
      float4 v = gsrc[idx4];
      int k  = idx4 >> 6;
      int n4 = idx4 & 63;
      *(float4*)&sF32[k * F32_STRIDE + n4 * 4] = v;
      atomicAdd(&sSumSq[k], v.x*v.x + v.y*v.y + v.z*v.z + v.w*v.w);
    }
    if (tid < NB) {
      int lb = labels[b * NB + tid] & (NCLS - 1);
      sLab[tid] = lb;
      atomicAdd(&sCnt[lb], 1.0f);
    }
#endif
    __syncthreads();
    if (tid < NB) sInv[tid] = 1.0f / fmaxf(sqrtf(sSumSq[tid]), CEPS);
    __syncthreads();

    // --- phase B1: normalize + f16 transpose, packed 8 halves per ds_store_b128 ---
    #pragma unroll
    for (int i = 0; i < (DD * NB / 8) / BLOCK; ++i) {   // 4 tasks/thread
      int task = tid + i * BLOCK;
      int n  = task & (DD - 1);
      int k0 = (task >> 8) * 8;
      v8h pk;
      #pragma unroll
      for (int j = 0; j < 8; ++j)
        pk[j] = (_Float16)(sF32[(k0 + j) * F32_STRIDE + n] * sInv[k0 + j]);
      *(v8h*)&sT[n * TR_STRIDE + k0] = pk;
    }

    // --- phase B2: build shared one-hot A fragments (contiguous k = 16*half + s per lane, same order as B) ---
    {
      int row   = tid >> 1;             // (kstep,cg,lane): 2*4*32 = 256 rows
      int part  = tid & 1;              // low/high 8 halves
      int kstep = row >> 7;
      int cg    = (row >> 5) & 3;
      int alane = row & 31;
      int cls   = cg * 16 + (alane & 15);
      int kadd  = kstep * 32 + (part << 3) + ((alane & 16) ? 16 : 0);
      v8h pk;
      #pragma unroll
      for (int j = 0; j < 8; ++j)
        pk[j] = (sLab[kadd + j] == cls) ? (_Float16)1.0f : (_Float16)0.0f;
      *(v8h*)&sA[row * 16 + part * 8] = pk;
    }
    __syncthreads();

    // --- phase C: WMMA one-hot segment-sum: D[16cls x 16dim] += A[16x32] * B[32x16] ---
    #pragma unroll
    for (int kstep = 0; kstep < 2; ++kstep) {
      const v16h bfrag =
          *(const v16h*)&sT[(w * 16 + m) * TR_STRIDE + kstep * 32 + (hi << 4)];
      #pragma unroll
      for (int cg = 0; cg < 4; ++cg) {
        const v16h afrag = *(const v16h*)&sA[((kstep * 4 + cg) * 32 + lane) * 16];
        acc[cg] = __builtin_amdgcn_wmma_f32_16x16x32_f16(
            false, afrag, false, bfrag, (short)0, acc[cg], false, false);
      }
    }
  }

  // --- flush: accumulators -> contention-spread partial buffers ---
  float* P = partials + (size_t)(blockIdx.x % numPartials) * (NCLS * DD);
  #pragma unroll
  for (int cg = 0; cg < 4; ++cg) {
    #pragma unroll
    for (int r = 0; r < 8; ++r) {
      int cls = cg * 16 + r + (hi << 3);   // C/D layout: VGPR r -> M=r (+8 for hi lanes)
      int col = w * 16 + m;
      atomicAdd(&P[cls * DD + col], acc[cg][r]);
    }
  }
  __syncthreads();
  if (tid < NCLS) atomicAdd(&counts[tid], sCnt[tid]);
}

// ---------------- reduce partial sums ----------------
__global__ void reduce_partials_kernel(const float* __restrict__ partials,
                                       float* __restrict__ Sfinal, int numPartials) {
  int e = blockIdx.x * blockDim.x + threadIdx.x;   // 0..16383
  float s = 0.0f;
  for (int p = 0; p < numPartials; ++p) s += partials[(size_t)p * (NCLS * DD) + e];
  Sfinal[e] = s;
}

// ---------------- finalize: loss = mean_{n_c>1} (1 - ||S_c||/n_c) ----------------
__global__ void finalize_kernel(const float* __restrict__ Sfinal,
                                const float* __restrict__ counts,
                                float* __restrict__ out) {
  __shared__ float redL[NCLS];
  __shared__ float redV[NCLS];
  int c = threadIdx.x;          // 64 threads
  float ss = 0.0f;
  for (int n = 0; n < DD; ++n) { float v = Sfinal[c * DD + n]; ss += v * v; }
  float cnt = counts[c];
  float loss_c = 0.0f, valid = 0.0f;
  if (cnt > 1.0f) {
    float normS  = sqrtf(ss);
    float u      = normS / fmaxf(cnt, 1.0f);                 // ||center before renorm||
    float simsum = (normS * u) / fmaxf(u, CEPS);             // = ||S|| with eps guard
    loss_c = (cnt - simsum) / fmaxf(cnt, 1.0f);
    valid  = 1.0f;
  }
  redL[c] = loss_c;
  redV[c] = valid;
  __syncthreads();
  if (c == 0) {
    float L = 0.0f, V = 0.0f;
    for (int i = 0; i < NCLS; ++i) { L += redL[i]; V += redV[i]; }
    out[0] = (V > 0.0f) ? (L / fmaxf(V, 1.0f)) : 0.0f;
  }
}

extern "C" void kernel_launch(void* const* d_in, const int* in_sizes, int n_in,
                              void* d_out, int out_size, void* d_ws, size_t ws_size,
                              hipStream_t stream) {
  const float4* emb    = (const float4*)d_in[0];
  const int*    labels = (const int*)d_in[1];   // jax default: int32 labels
  // d_in[2] = epoch (5 >= START) -> always compute, matching traced reference
  int N = in_sizes[0] / DD;

  float* ws = (float*)d_ws;
  const size_t slot = (size_t)NCLS * DD;        // 16384 floats
  long avail = (long)(ws_size / 4) - (long)slot - 64;  // reserve Sfinal + counts
  int numPartials = (int)(avail / (long)slot);
  if (numPartials < 1)  numPartials = 1;
  if (numPartials > 64) numPartials = 64;

  float* partials = ws;
  float* counts   = ws + (size_t)numPartials * slot;
  float* Sfinal   = counts + NCLS;

  int zeroN = numPartials * (int)slot + NCLS;
  zero_ws_kernel<<<(zeroN + 255) / 256, 256, 0, stream>>>(ws, zeroN);

  int numBatches = N / NB;
  int grid = numBatches < 512 ? numBatches : 512;
  compact_main_kernel<<<grid, BLOCK, 0, stream>>>(emb, labels, partials, counts,
                                                  N, numPartials);

  reduce_partials_kernel<<<(NCLS * DD) / 256, 256, 0, stream>>>(partials, Sfinal,
                                                                numPartials);
  finalize_kernel<<<1, NCLS, 0, stream>>>(Sfinal, counts, (float*)d_out);
}
